// MetaEmbedding_Classifier_13322988552210
// MI455X (gfx1250) — compile-verified, matrix operand order checked by reading, not run on hardware
//
#include <hip/hip_runtime.h>
#include <hip/hip_bf16.h>

// ---------------------------------------------------------------------------
// MetaEmbedding classifier forward for MI455X (gfx1250, wave32, WMMA).
// All GEMMs use v_wmma_f32_16x16x32_bf16: weights are streamed from HBM in
// FP32 and converted to BF16 in registers (each weight byte touched once),
// M=128 rows = 8 WMMA tiles held in accumulators per wave.
// ---------------------------------------------------------------------------

typedef __bf16           bf16x16 __attribute__((ext_vector_type(16)));
typedef unsigned short   u16x8   __attribute__((ext_vector_type(8)));
typedef unsigned short   u16x16  __attribute__((ext_vector_type(16)));
typedef float            f32x8   __attribute__((ext_vector_type(8)));

#define BATCH 128
#define FEAT  2048
#define NCLS  1000
#define NCLSP 1024   // padded K for probs @ centroids

__device__ __forceinline__ unsigned short f2bf(float f) {
    unsigned int u = __float_as_uint(f);
    unsigned int r = (u + 0x7FFFu + ((u >> 16) & 1u)) >> 16;   // RNE
    return (unsigned short)r;
}

// ---------------------------------------------------------------------------
// out[128,N] = A_bf16[128,K] @ W[N,K]^T (+ bias[n]).   One wave = 16 N-cols
// across all 128 M-rows (8 accumulator tiles). Block = 8 waves = 128 N-cols.
// ---------------------------------------------------------------------------
__global__ __launch_bounds__(256)
void gemm_ABt_wmma(const unsigned short* __restrict__ A,
                   const float* __restrict__ W,
                   const float* __restrict__ bias,
                   float* __restrict__ out, int N, int K) {
    const int lane = threadIdx.x & 31;
    const int wave = threadIdx.x >> 5;
    const int lm   = lane & 15;
    const int lh   = lane >> 4;
    const int ncol = blockIdx.x * 128 + wave * 16 + lm;
    const int ncl  = ncol < N ? ncol : (N - 1);

    f32x8 acc[8];
#pragma unroll
    for (int t = 0; t < 8; ++t)
#pragma unroll
        for (int v = 0; v < 8; ++v) acc[t][v] = 0.0f;

    const float* wrow = W + (size_t)ncl * K;

    for (int kb = 0; kb < K; kb += 32) {
        const int kOff = kb + lh * 16;
        // B fragment: 16 contiguous FP32 weights -> BF16
        u16x16 bu;
        const float* wp = wrow + kOff;
#pragma unroll
        for (int i = 0; i < 16; ++i) bu[i] = f2bf(wp[i]);
        bf16x16 bF = __builtin_bit_cast(bf16x16, bu);

#pragma unroll
        for (int mt = 0; mt < 8; ++mt) {
            const unsigned short* ap = A + (size_t)(mt * 16 + lm) * K + kOff;
            u16x8 a0 = *(const u16x8*)ap;
            u16x8 a1 = *(const u16x8*)(ap + 8);
            u16x16 au;
#pragma unroll
            for (int i = 0; i < 8; ++i) { au[i] = a0[i]; au[i + 8] = a1[i]; }
            bf16x16 aF = __builtin_bit_cast(bf16x16, au);
            acc[mt] = __builtin_amdgcn_wmma_f32_16x16x32_bf16(
                false, aF, false, bF, (short)0, acc[mt], false, false);
        }
    }

    const float bv = bias ? bias[ncl] : 0.0f;
    if (ncol < N) {
#pragma unroll
        for (int mt = 0; mt < 8; ++mt)
#pragma unroll
            for (int v = 0; v < 8; ++v) {
                int row = mt * 16 + v + 8 * lh;
                out[(size_t)row * N + ncol] = acc[mt][v] + bv;
            }
    }
}

// ---------------------------------------------------------------------------
// out[128,N] = A_bf16[128,Kpad] @ Bmat[Kreal,ldB]  (B row-major, K guarded)
// ---------------------------------------------------------------------------
__global__ __launch_bounds__(256)
void gemm_AB_wmma(const unsigned short* __restrict__ A,
                  const float* __restrict__ Bmat,
                  float* __restrict__ out,
                  int N, int Kpad, int Kreal, int ldB) {
    const int lane = threadIdx.x & 31;
    const int wave = threadIdx.x >> 5;
    const int lm   = lane & 15;
    const int lh   = lane >> 4;
    const int ncol = blockIdx.x * 128 + wave * 16 + lm;
    const int ncl  = ncol < N ? ncol : (N - 1);

    f32x8 acc[8];
#pragma unroll
    for (int t = 0; t < 8; ++t)
#pragma unroll
        for (int v = 0; v < 8; ++v) acc[t][v] = 0.0f;

    for (int kb = 0; kb < Kpad; kb += 32) {
        const int kOff = kb + lh * 16;
        u16x16 bu;
#pragma unroll
        for (int i = 0; i < 16; ++i) {
            int k = kOff + i;
            bu[i] = (k < Kreal) ? f2bf(Bmat[(size_t)k * ldB + ncl]) : (unsigned short)0;
        }
        bf16x16 bF = __builtin_bit_cast(bf16x16, bu);

#pragma unroll
        for (int mt = 0; mt < 8; ++mt) {
            const unsigned short* ap = A + (size_t)(mt * 16 + lm) * Kpad + kOff;
            u16x8 a0 = *(const u16x8*)ap;
            u16x8 a1 = *(const u16x8*)(ap + 8);
            u16x16 au;
#pragma unroll
            for (int i = 0; i < 8; ++i) { au[i] = a0[i]; au[i + 8] = a1[i]; }
            bf16x16 aF = __builtin_bit_cast(bf16x16, au);
            acc[mt] = __builtin_amdgcn_wmma_f32_16x16x32_bf16(
                false, aF, false, bF, (short)0, acc[mt], false, false);
        }
    }

    if (ncol < N) {
#pragma unroll
        for (int mt = 0; mt < 8; ++mt)
#pragma unroll
            for (int v = 0; v < 8; ++v) {
                int row = mt * 16 + v + 8 * lh;
                out[(size_t)row * N + ncol] = acc[mt][v];
            }
    }
}

// ---------------------------------------------------------------------------
// Small helper kernels
// ---------------------------------------------------------------------------
__global__ __launch_bounds__(256)
void cvt_bf16_kernel(const float* __restrict__ in, unsigned short* __restrict__ out, int n) {
    int idx = blockIdx.x * 256 + threadIdx.x;
    if (idx < n) out[idx] = f2bf(in[idx]);
}

__global__ __launch_bounds__(256)
void rownorm2_kernel(const float* __restrict__ X, int D, float* __restrict__ out) {
    __shared__ float sm[256];
    const int row = blockIdx.x;
    const float* r = X + (size_t)row * D;
    float s = 0.0f;
    for (int j = threadIdx.x; j < D; j += 256) { float v = r[j]; s += v * v; }
    sm[threadIdx.x] = s; __syncthreads();
    for (int st = 128; st > 0; st >>= 1) {
        if (threadIdx.x < st) sm[threadIdx.x] += sm[threadIdx.x + st];
        __syncthreads();
    }
    if (threadIdx.x == 0) out[row] = sm[0];
}

__global__ __launch_bounds__(256)
void reach_kernel(const float* __restrict__ xc, const float* __restrict__ xx,
                  const float* __restrict__ cc, float* __restrict__ reach, int C) {
    __shared__ float sm[256];
    const int b = blockIdx.x;
    const float xb = xx[b];
    float m = 3.4e38f;
    for (int j = threadIdx.x; j < C; j += 256) {
        float d2 = xb + cc[j] - 2.0f * xc[(size_t)b * C + j];
        m = fminf(m, d2);
    }
    sm[threadIdx.x] = m; __syncthreads();
    for (int st = 128; st > 0; st >>= 1) {
        if (threadIdx.x < st) sm[threadIdx.x] = fminf(sm[threadIdx.x], sm[threadIdx.x + st]);
        __syncthreads();
    }
    if (threadIdx.x == 0) reach[b] = 10.0f * rsqrtf(fmaxf(sm[0], 1e-12f));
}

__global__ __launch_bounds__(256)
void softmax_kernel(const float* __restrict__ logits, unsigned short* __restrict__ probs,
                    int C, int Cp) {
    __shared__ float sm[256];
    const int b = blockIdx.x;
    const float* row = logits + (size_t)b * C;
    float m = -3.4e38f;
    for (int j = threadIdx.x; j < C; j += 256) m = fmaxf(m, row[j]);
    sm[threadIdx.x] = m; __syncthreads();
    for (int st = 128; st > 0; st >>= 1) {
        if (threadIdx.x < st) sm[threadIdx.x] = fmaxf(sm[threadIdx.x], sm[threadIdx.x + st]);
        __syncthreads();
    }
    const float mx = sm[0]; __syncthreads();
    float s = 0.0f;
    for (int j = threadIdx.x; j < C; j += 256) s += __expf(row[j] - mx);
    sm[threadIdx.x] = s; __syncthreads();
    for (int st = 128; st > 0; st >>= 1) {
        if (threadIdx.x < st) sm[threadIdx.x] += sm[threadIdx.x + st];
        __syncthreads();
    }
    const float inv = 1.0f / sm[0];
    for (int j = threadIdx.x; j < Cp; j += 256)
        probs[(size_t)b * Cp + j] =
            (j < C) ? f2bf(__expf(row[j] - mx) * inv) : (unsigned short)0;
}

__global__ __launch_bounds__(256)
void finalize_kernel(const float* __restrict__ x, const float* __restrict__ selpre,
                     const float* __restrict__ mem, const float* __restrict__ reach,
                     float* __restrict__ out_direct, float* __restrict__ out_infused,
                     float* __restrict__ feat, unsigned short* __restrict__ featb, int n) {
    int idx = blockIdx.x * 256 + threadIdx.x;
    if (idx >= n) return;
    int b = idx >> 11;                       // D = 2048
    float xv  = x[idx];
    float inf = tanhf(selpre[idx]) * mem[idx];
    float f   = reach[b] * (xv + inf);
    out_direct[idx]  = xv;
    out_infused[idx] = inf;
    feat[idx]  = f;
    featb[idx] = f2bf(f);
}

__global__ __launch_bounds__(256)
void logits_kernel(const float* __restrict__ fw, const float* __restrict__ featn2,
                   const float* __restrict__ wcn2, float* __restrict__ out, int C, int n) {
    int idx = blockIdx.x * 256 + threadIdx.x;
    if (idx >= n) return;
    int b = idx / C, j = idx - b * C;
    float nrm = sqrtf(featn2[b]);
    out[idx] = 16.0f * fw[idx] / ((1.0f + nrm) * sqrtf(fmaxf(wcn2[j], 1e-24f)));
}

// ---------------------------------------------------------------------------
extern "C" void kernel_launch(void* const* d_in, const int* in_sizes, int n_in,
                              void* d_out, int out_size, void* d_ws, size_t ws_size,
                              hipStream_t stream) {
    const float* x     = (const float*)d_in[0];
    const float* cent  = (const float*)d_in[1];
    const float* Whall = (const float*)d_in[2];
    const float* bhall = (const float*)d_in[3];
    const float* Wsel  = (const float*)d_in[4];
    const float* bsel  = (const float*)d_in[5];
    const float* Wcos  = (const float*)d_in[6];
    float* out = (float*)d_out;

    float* out_logits  = out;                                   // [128,1000]
    float* out_direct  = out + BATCH * NCLS;                    // [128,2048]
    float* out_infused = out_direct + BATCH * FEAT;             // [128,2048]

    char* ws = (char*)d_ws;
    size_t off = 0;
    unsigned short* x_bf    = (unsigned short*)(ws + off); off += (size_t)BATCH * FEAT * 2;   // 512K
    unsigned short* feat_bf = (unsigned short*)(ws + off); off += (size_t)BATCH * FEAT * 2;   // 512K
    unsigned short* probs   = (unsigned short*)(ws + off); off += (size_t)BATCH * NCLSP * 2;  // 256K
    float* xc     = (float*)(ws + off); off += (size_t)BATCH * NCLS * 4;
    float* hall   = (float*)(ws + off); off += (size_t)BATCH * NCLS * 4;
    float* fw     = (float*)(ws + off); off += (size_t)BATCH * NCLS * 4;
    float* selpre = (float*)(ws + off); off += (size_t)BATCH * FEAT * 4;
    float* mem    = (float*)(ws + off); off += (size_t)BATCH * FEAT * 4;
    float* feat   = (float*)(ws + off); off += (size_t)BATCH * FEAT * 4;
    float* xx     = (float*)(ws + off); off += 512;
    float* featn2 = (float*)(ws + off); off += 512;
    float* reach  = (float*)(ws + off); off += 512;
    float* cc     = (float*)(ws + off); off += (size_t)NCLS * 4;
    float* wcn2   = (float*)(ws + off); off += (size_t)NCLS * 4;

    const int nXD = BATCH * FEAT;   // 262144

    // Stage 0: x -> bf16 ; row norms
    cvt_bf16_kernel<<<nXD / 256, 256, 0, stream>>>(x, x_bf, nXD);
    rownorm2_kernel<<<BATCH, 256, 0, stream>>>(x, FEAT, xx);
    rownorm2_kernel<<<NCLS, 256, 0, stream>>>(cent, FEAT, cc);
    rownorm2_kernel<<<NCLS, 256, 0, stream>>>(Wcos, FEAT, wcn2);

    // Stage 1: the three x-GEMMs (weights streamed once, WMMA bf16)
    gemm_ABt_wmma<<<(NCLS + 127) / 128, 256, 0, stream>>>(x_bf, cent, nullptr, xc, NCLS, FEAT);
    gemm_ABt_wmma<<<(NCLS + 127) / 128, 256, 0, stream>>>(x_bf, Whall, bhall, hall, NCLS, FEAT);
    gemm_ABt_wmma<<<FEAT / 128, 256, 0, stream>>>(x_bf, Wsel, bsel, selpre, FEAT, FEAT);

    // Stage 2: reachability + softmax attention
    reach_kernel<<<BATCH, 256, 0, stream>>>(xc, xx, cc, reach, NCLS);
    softmax_kernel<<<BATCH, 256, 0, stream>>>(hall, probs, NCLS, NCLSP);

    // Stage 3: memory_feature = probs @ centroids (WMMA, row-major B)
    gemm_AB_wmma<<<FEAT / 128, 256, 0, stream>>>(probs, cent, mem, FEAT, NCLSP, NCLS, FEAT);

    // Stage 4: fuse tanh/selector/reachability, emit direct+infused outputs
    finalize_kernel<<<nXD / 256, 256, 0, stream>>>(x, selpre, mem, reach,
                                                   out_direct, out_infused, feat, feat_bf, nXD);
    rownorm2_kernel<<<BATCH, 256, 0, stream>>>(feat, FEAT, featn2);

    // Stage 5: cosnorm logits
    gemm_ABt_wmma<<<(NCLS + 127) / 128, 256, 0, stream>>>(feat_bf, Wcos, nullptr, fw, NCLS, FEAT);
    logits_kernel<<<(BATCH * NCLS) / 256, 256, 0, stream>>>(fw, featn2, wcn2, out_logits,
                                                            NCLS, BATCH * NCLS);
}
